// Seq2SeqOutputLayer_68101001445959
// MI455X (gfx1250) — matrix operand order repaired and checked by reading, not compiled, not run on hardware
//
#include <hip/hip_runtime.h>
#include <math.h>

typedef __bf16 bf16;
typedef __attribute__((ext_vector_type(16))) __bf16 v16bf;
typedef __attribute__((ext_vector_type(8)))  __bf16 v8bf;
typedef __attribute__((ext_vector_type(8)))  float  v8f;

static constexpr int B = 64, S = 128, T = 64, D = 512, V = 32000, G = 3 * D; // G=1536

// ---------------------------------------------------------------------------
// Generic C[64,N] = A[64,K] * Bstore[N,K]^T (+bias), bf16 inputs, f32 out,
// optional bf16 mirror of the output. One block = 64x64 tile, 4 waves,
// each wave computes a 16(M) x 64(N) strip with 4 f32 accumulators and
// reuses its A fragment across 4 v_wmma_f32_16x16x32_bf16 per K-step.
// A (64 x K bf16) is staged in LDS once (ds_load_b128 fragment reads).
// ---------------------------------------------------------------------------
__global__ __launch_bounds__(128)
void gemm_abT_wmma(const bf16* __restrict__ A, long a_row_stride,
                   const bf16* __restrict__ Bm, int K,
                   const float* __restrict__ bias,
                   float* __restrict__ C, long c_row_stride,
                   bf16* __restrict__ Cbf, long cbf_row_stride)
{
    extern __shared__ char smem_raw[];
    bf16* As = (bf16*)smem_raw;                 // 64 x K bf16

    const int tid  = threadIdx.x;
    const int lane = tid & 31;
    const int wave = tid >> 5;
    const long n0  = (long)blockIdx.x * 64;

    // Stage A into LDS: 16-byte vector copies
    const int total8 = (64 * K) / 8;
    for (int i = tid; i < total8; i += 128) {
        const int eb = i * 8;
        const int m  = eb / K;
        const int k  = eb % K;
        *(v8bf*)&As[m * K + k] = *(const v8bf*)&A[(long)m * a_row_stride + k];
    }
    __syncthreads();

    v8f acc[4] = {};

    const int m0   = wave * 16;
    const int mrow = m0 + (lane & 15);
    const int grp  = lane >> 4;
    const int ncol = lane & 15;

    for (int k0 = 0; k0 < K; k0 += 32) {
        // A fragment: lane holds row mrow, K = k0 + grp*16 .. +15 (contiguous)
        const v8bf alo = *(const v8bf*)&As[mrow * K + k0 + grp * 16];
        const v8bf ahi = *(const v8bf*)&As[mrow * K + k0 + grp * 16 + 8];
        const v16bf afrag = __builtin_shufflevector(
            alo, ahi, 0, 1, 2, 3, 4, 5, 6, 7, 8, 9, 10, 11, 12, 13, 14, 15);

        // Prefetch next K block of B for this lane's first column strip
        if (k0 + 32 < K)
            __builtin_prefetch(&Bm[(n0 + ncol) * (long)K + k0 + 32 + grp * 16], 0, 1);

#pragma unroll
        for (int nt = 0; nt < 4; ++nt) {
            const long n = n0 + nt * 16 + ncol;
            // B fragment: lane holds col n, K = k0 + grp*16 .. +15 (contiguous)
            const v16bf bfrag = *(const v16bf*)&Bm[n * (long)K + k0 + grp * 16];
            acc[nt] = __builtin_amdgcn_wmma_f32_16x16x32_bf16(
                false, afrag, false, bfrag, (short)0, acc[nt], false, false);
        }
    }

    // D layout: lane ncol = N, VGPR r -> M = m0 + r + 8*grp
#pragma unroll
    for (int nt = 0; nt < 4; ++nt) {
        const long n  = n0 + nt * 16 + ncol;
        const float bv = bias ? bias[n] : 0.0f;
#pragma unroll
        for (int r = 0; r < 8; ++r) {
            const int   m = m0 + r + 8 * grp;
            const float v = acc[nt][r] + bv;
            C[(long)m * c_row_stride + n] = v;
            if (Cbf) Cbf[(long)m * cbf_row_stride + n] = (bf16)v;
        }
    }
}

// ---------------------------------------------------------------------------
// Elementwise GRU gate combine: h' = (1-z)*n + z*h ; gates ordered (r,z,n).
// Optionally writes bf16 mirror and an enc_states slot.
// ---------------------------------------------------------------------------
__global__ __launch_bounds__(256)
void gru_elem(const float* __restrict__ gi, const float* __restrict__ gh,
              const float* __restrict__ h,
              float* __restrict__ hout, bf16* __restrict__ hout_bf,
              float* __restrict__ enc_slot, long enc_row_stride)
{
    const int tid = blockIdx.x * blockDim.x + threadIdx.x;
    if (tid >= B * D) return;
    const int b = tid >> 9;       // /D
    const int d = tid & (D - 1);

    const float ir = gi[b * G + d],      hr = gh[b * G + d];
    const float iz = gi[b * G + D + d],  hz = gh[b * G + D + d];
    const float in = gi[b * G + 2*D + d], hn = gh[b * G + 2*D + d];
    const float hv = h[tid];

    const float r = 1.0f / (1.0f + __expf(-(ir + hr)));
    const float z = 1.0f / (1.0f + __expf(-(iz + hz)));
    const float n = tanhf(in + r * hn);
    const float o = (1.0f - z) * n + z * hv;

    hout[tid] = o;
    if (hout_bf) hout_bf[tid] = (bf16)o;
    if (enc_slot) enc_slot[(long)b * enc_row_stride + d] = o;
}

// ---------------------------------------------------------------------------
// Per-step attention: scores = enc_states[b]·h[b], softmax over S, ctx,
// and emit cat = [h, ctx] in bf16 (input to the combine GEMM).
// One block per batch row; h and scores live in LDS.
// ---------------------------------------------------------------------------
__global__ __launch_bounds__(128)
void attention_ctx(const float* __restrict__ enc, const float* __restrict__ hd,
                   bf16* __restrict__ cat)
{
    __shared__ float hs[D];
    __shared__ float sc[S];
    __shared__ float red[S];

    const int b = blockIdx.x;
    const int t = threadIdx.x;                  // 0..127 == S
    const float* encb = enc + (long)b * S * D;

    for (int d = t; d < D; d += 128) hs[d] = hd[b * D + d];
    __syncthreads();

    // score for source position t
    {
        const float* row = encb + t * D;
        float acc = 0.0f;
        for (int d = 0; d < D; d += 4) {
            const float4 e = *(const float4*)&row[d];
            acc += e.x * hs[d] + e.y * hs[d + 1] + e.z * hs[d + 2] + e.w * hs[d + 3];
        }
        sc[t] = acc;
    }
    __syncthreads();

    // softmax: max then sum
    red[t] = sc[t];
    __syncthreads();
    for (int off = 64; off > 0; off >>= 1) {
        if (t < off) red[t] = fmaxf(red[t], red[t + off]);
        __syncthreads();
    }
    const float mx = red[0];
    __syncthreads();
    const float ev = __expf(sc[t] - mx);
    red[t] = ev;
    __syncthreads();
    for (int off = 64; off > 0; off >>= 1) {
        if (t < off) red[t] += red[t + off];
        __syncthreads();
    }
    const float inv = 1.0f / red[0];
    __syncthreads();
    sc[t] = ev * inv;
    __syncthreads();

    // ctx[d] = sum_s p[s] * enc[b,s,d]; also write h half of cat
    for (int d = t; d < D; d += 128) {
        float c = 0.0f;
        for (int s = 0; s < S; ++s) c += sc[s] * encb[s * D + d];
        cat[(long)b * 2 * D + d]     = (bf16)hs[d];
        cat[(long)b * 2 * D + D + d] = (bf16)c;
    }
}

// ---------------------------------------------------------------------------
// Small utility kernels
// ---------------------------------------------------------------------------
__global__ __launch_bounds__(256)
void conv_f32_bf16(const float* __restrict__ src, bf16* __restrict__ dst, long n)
{
    const long i = (long)blockIdx.x * blockDim.x + threadIdx.x;
    if (i < n) dst[i] = (bf16)src[i];
}

__global__ __launch_bounds__(256)
void copy_f32_and_bf16(const float* __restrict__ src, float* __restrict__ dstf,
                       bf16* __restrict__ dstb, long n)
{
    const long i = (long)blockIdx.x * blockDim.x + threadIdx.x;
    if (i < n) { const float v = src[i]; dstf[i] = v; dstb[i] = (bf16)v; }
}

__global__ __launch_bounds__(256)
void embed_gather_bf16(const int* __restrict__ y_id, const float* __restrict__ embed_W,
                       bf16* __restrict__ y_bf)
{
    const long i = (long)blockIdx.x * blockDim.x + threadIdx.x; // B*T*D
    if (i >= (long)B * T * D) return;
    const int  d  = (int)(i & (D - 1));
    const long bt = i >> 9;
    const int  v  = y_id[bt];
    y_bf[i] = (bf16)embed_W[(long)v * D + d];
}

// ---------------------------------------------------------------------------
// Host orchestration (graph-capture safe: only kernel launches on `stream`)
// ---------------------------------------------------------------------------
extern "C" void kernel_launch(void* const* d_in, const int* in_sizes, int n_in,
                              void* d_out, int out_size, void* d_ws, size_t ws_size,
                              hipStream_t stream)
{
    const float* x        = (const float*)d_in[0];
    const float* q        = (const float*)d_in[1];
    const int*   y_id     = (const int*)  d_in[2];
    const float* embed_W  = (const float*)d_in[3];
    const float* enc_Wih  = (const float*)d_in[4];
    const float* enc_Whh  = (const float*)d_in[5];
    const float* enc_bih  = (const float*)d_in[6];
    const float* enc_bhh  = (const float*)d_in[7];
    const float* dec_Wih  = (const float*)d_in[8];
    const float* dec_Whh  = (const float*)d_in[9];
    const float* dec_bih  = (const float*)d_in[10];
    const float* dec_bhh  = (const float*)d_in[11];
    const float* comb_W   = (const float*)d_in[12];
    const float* comb_b   = (const float*)d_in[13];
    float* out = (float*)d_out;

    // workspace carve-out
    char* p = (char*)d_ws;
    auto carve = [&](size_t bytes) -> char* {
        char* r = p; p += (bytes + 255) & ~(size_t)255; return r;
    };
    bf16* embed_bf   = (bf16*)carve((size_t)V * D * 2);
    bf16* x_bf       = (bf16*)carve((size_t)B * S * D * 2);
    bf16* y_bf       = (bf16*)carve((size_t)B * T * D * 2);
    bf16* enc_Wih_bf = (bf16*)carve((size_t)G * D * 2);
    bf16* enc_Whh_bf = (bf16*)carve((size_t)G * D * 2);
    bf16* dec_Wih_bf = (bf16*)carve((size_t)G * D * 2);
    bf16* dec_Whh_bf = (bf16*)carve((size_t)G * D * 2);
    bf16* comb_W_bf  = (bf16*)carve((size_t)D * 2 * D * 2);
    float* enc_f32   = (float*)carve((size_t)B * S * D * 4);
    float* h_f32     = (float*)carve((size_t)B * D * 4);
    bf16*  h_bf      = (bf16*)carve((size_t)B * D * 2);
    float* stt_f32   = (float*)carve((size_t)B * D * 4);
    bf16*  stt_bf    = (bf16*)carve((size_t)B * D * 2);
    float* hd_f32    = (float*)carve((size_t)B * D * 4);
    float* gi_ws     = (float*)carve((size_t)B * G * 4);
    float* gh_ws     = (float*)carve((size_t)B * G * 4);
    bf16*  cat_bf    = (bf16*)carve((size_t)B * 2 * D * 2);

    auto blk = [](long n, int t) { return (unsigned)((n + t - 1) / t); };

    // one-time conversions
    conv_f32_bf16<<<blk((long)V * D, 256), 256, 0, stream>>>(embed_W, embed_bf, (long)V * D);
    conv_f32_bf16<<<blk((long)B * S * D, 256), 256, 0, stream>>>(x, x_bf, (long)B * S * D);
    conv_f32_bf16<<<blk((long)G * D, 256), 256, 0, stream>>>(enc_Wih, enc_Wih_bf, (long)G * D);
    conv_f32_bf16<<<blk((long)G * D, 256), 256, 0, stream>>>(enc_Whh, enc_Whh_bf, (long)G * D);
    conv_f32_bf16<<<blk((long)G * D, 256), 256, 0, stream>>>(dec_Wih, dec_Wih_bf, (long)G * D);
    conv_f32_bf16<<<blk((long)G * D, 256), 256, 0, stream>>>(dec_Whh, dec_Whh_bf, (long)G * D);
    conv_f32_bf16<<<blk((long)D * 2 * D, 256), 256, 0, stream>>>(comb_W, comb_W_bf, (long)D * 2 * D);
    embed_gather_bf16<<<blk((long)B * T * D, 256), 256, 0, stream>>>(y_id, embed_W, y_bf);
    copy_f32_and_bf16<<<blk((long)B * D, 256), 256, 0, stream>>>(q, h_f32, h_bf, (long)B * D);

    const size_t ldsA_512  = (size_t)64 * 512 * 2;   // 64 KB
    const size_t ldsA_1024 = (size_t)64 * 1024 * 2;  // 128 KB

    // ---------------- Encoder: 128 sequential GRU steps ----------------
    for (int t = 0; t < S; ++t) {
        gemm_abT_wmma<<<G / 64, 128, ldsA_512, stream>>>(
            x_bf + (long)t * D, (long)S * D, enc_Wih_bf, D, enc_bih,
            gi_ws, G, nullptr, 0);
        gemm_abT_wmma<<<G / 64, 128, ldsA_512, stream>>>(
            h_bf, D, enc_Whh_bf, D, enc_bhh,
            gh_ws, G, nullptr, 0);
        gru_elem<<<blk((long)B * D, 256), 256, 0, stream>>>(
            gi_ws, gh_ws, h_f32, h_f32, h_bf,
            enc_f32 + (long)t * D, (long)S * D);
    }

    // decoder initial state = h_last
    copy_f32_and_bf16<<<blk((long)B * D, 256), 256, 0, stream>>>(h_f32, stt_f32, stt_bf, (long)B * D);

    // ---------------- Decoder: 64 sequential steps ----------------
    for (int t = 0; t < T; ++t) {
        gemm_abT_wmma<<<G / 64, 128, ldsA_512, stream>>>(
            y_bf + (long)t * D, (long)T * D, dec_Wih_bf, D, dec_bih,
            gi_ws, G, nullptr, 0);
        gemm_abT_wmma<<<G / 64, 128, ldsA_512, stream>>>(
            stt_bf, D, dec_Whh_bf, D, dec_bhh,
            gh_ws, G, nullptr, 0);
        gru_elem<<<blk((long)B * D, 256), 256, 0, stream>>>(
            gi_ws, gh_ws, stt_f32, hd_f32, nullptr, nullptr, 0);
        attention_ctx<<<B, 128, 0, stream>>>(enc_f32, hd_f32, cat_bf);
        gemm_abT_wmma<<<D / 64, 128, ldsA_1024, stream>>>(
            cat_bf, 2 * D, comb_W_bf, 2 * D, comb_b,
            stt_f32, D, stt_bf, D);
        // logits: [64,512] x [32000,512]^T, written straight into d_out[:, t, :]
        gemm_abT_wmma<<<V / 64, 128, ldsA_512, stream>>>(
            stt_bf, D, embed_bf, D, nullptr,
            out + (long)t * V, (long)T * V, nullptr, 0);
    }
}
